// MultiPerspectiveNN_77292231459200
// MI455X (gfx1250) — matrix operand order repaired and checked by reading, not compiled, not run on hardware
//
#include <hip/hip_runtime.h>
#include <hip/hip_bf16.h>

// ---------------------------------------------------------------------------
// LSTM (T=512, B=256, D=512, H=512) for MI455X / gfx1250.
//   - persistent kernel, one launch for the full recurrence
//   - f16 WMMA 16x16x32, fp32 accumulation
//   - 512 waves, each owns one 16x16 (batch x hidden) tile and all 4 gates
//   - weights pre-packed into per-lane WMMA B-fragment order (f16)
//   - grid-wide barrier via device-scope atomics between timesteps
// ---------------------------------------------------------------------------

typedef __attribute__((ext_vector_type(16))) _Float16 v16h;
typedef __attribute__((ext_vector_type(8)))  _Float16 v8h;
typedef __attribute__((ext_vector_type(8)))  float    v8f;

#define T_STEPS 512
#define B_SZ    256
#define D_SZ    512
#define H_SZ    512
#define NBLOCKS 64
#define NTHREADS 256   // 8 waves of 32

// workspace layout (bytes)
#define WS_WPACK_OFF 0u                      // 4 MiB packed f16 weights
#define WS_BIAS_OFF  (4u << 20)              // 8 KiB  f32 bias [4][512]
#define WS_HBUF_OFF  ((4u << 20) + 8192u)    // 2 x 256 KiB f16 h ping-pong
#define WS_BAR_OFF   ((4u << 20) + 8192u + 2u * 262144u)  // barrier counter

// A-fragment (16x32 f16, ISA 7.12.2): lane l holds row (l&15); half h maps to
//   K = (h<8 ? h : h+8) + (l>=16 ? 8 : 0)
// -> from row-major storage: two contiguous 8-half chunks at k0 and k0+16,
//    k0 = kb*32 + (l>>4)*8.
// B-fragment (32x16 f16): lane l holds col (l&15); half h maps to
//   K = (h<8 ? h : h+8) + (l>=16 ? 8 : 0), the same order as A
// We pre-pack weights so each lane reads its 16 halves contiguously.

__global__ void pack_weights_kernel(
    const float* Wxi, const float* Whi, const float* bi,
    const float* Wxf, const float* Whf, const float* bf,
    const float* Wxo, const float* Who, const float* bo,
    const float* Wxc, const float* Whc, const float* bc,
    _Float16* wpack, float* biasbuf) {
  const int idx = blockIdx.x * blockDim.x + threadIdx.x;
  const int NPACK = 4 * 32 * 32 * 32 * 16;  // g, kb, nt, lane, half
  if (idx < NPACK) {
    const int half = idx & 15;
    const int lane = (idx >> 4) & 31;
    const int nt   = (idx >> 9) & 31;
    const int kb   = (idx >> 14) & 31;
    const int g    = idx >> 19;
    const int n = nt * 16 + (lane & 15);
    const int k = kb * 32 + ((lane >> 4) * 8) + (half < 8 ? half : half + 8);  // 0..1023
    const float* Wx = (g == 0) ? Wxi : (g == 1) ? Wxf : (g == 2) ? Wxo : Wxc;
    const float* Wh = (g == 0) ? Whi : (g == 1) ? Whf : (g == 2) ? Who : Whc;
    const float w = (k < D_SZ) ? Wx[k * H_SZ + n] : Wh[(k - D_SZ) * H_SZ + n];
    wpack[idx] = (_Float16)w;
  }
  if (idx < 4 * H_SZ) {
    const int g = idx >> 9;
    const int n = idx & (H_SZ - 1);
    const float* bs = (g == 0) ? bi : (g == 1) ? bf : (g == 2) ? bo : bc;
    biasbuf[idx] = bs[n];
  }
}

__global__ void prep_state_kernel(const float* Hin, const float* Cin,
                                  _Float16* hbuf0, float* Cstate,
                                  unsigned* bar) {
  const int idx = blockIdx.x * blockDim.x + threadIdx.x;
  if (idx < B_SZ * H_SZ) {
    hbuf0[idx] = (_Float16)Hin[idx];
    Cstate[idx] = Cin[idx];
  }
  if (idx == 0) *bar = 0u;
}

__device__ __forceinline__ float fast_sigmoid(float x) {
  return 1.0f / (1.0f + __expf(-x));
}
__device__ __forceinline__ float fast_tanh(float x) {
  // 1 - 2/(e^{2x}+1): saturates correctly for |x| large (no inf/inf NaN)
  const float e = __expf(2.0f * x);
  return 1.0f - 2.0f / (e + 1.0f);
}

__device__ __forceinline__ void grid_barrier(unsigned* bar, unsigned step) {
  __threadfence();          // release our stores to agent scope
  __syncthreads();
  if (threadIdx.x == 0) {
    __hip_atomic_fetch_add(bar, 1u, __ATOMIC_ACQ_REL, __HIP_MEMORY_SCOPE_AGENT);
    const unsigned target = (step + 1u) * (unsigned)NBLOCKS;
    while (__hip_atomic_load(bar, __ATOMIC_ACQUIRE,
                             __HIP_MEMORY_SCOPE_AGENT) < target) {
      __builtin_amdgcn_s_sleep(1);
    }
  }
  __syncthreads();
  __threadfence();          // acquire other blocks' stores
}

__global__ __launch_bounds__(NTHREADS, 1)
void lstm_persistent_kernel(const float* __restrict__ X,
                            const _Float16* __restrict__ wpack,
                            const float* __restrict__ biasbuf,
                            _Float16* __restrict__ hbuf,   // 2 x [256][512]
                            float* __restrict__ Cstate,    // [256][512] (in d_out)
                            float* __restrict__ out,       // [T][256][512]
                            float* __restrict__ Hout,      // [256][512]
                            unsigned* __restrict__ bar) {
  const int lane = threadIdx.x & 31;
  const int wave = threadIdx.x >> 5;
  const int wid  = blockIdx.x * (NTHREADS / 32) + wave;  // 0..511
  const int bt   = wid >> 5;          // batch tile   0..15
  const int nt   = wid & 31;          // hidden tile  0..31
  const int lhi  = lane >> 4;         // 0/1 (lane half)
  const int rg   = bt * 16 + (lane & 15);  // batch row this lane loads (A)
  const int col  = nt * 16 + (lane & 15);  // output column this lane owns (C/D)
  const int rowbase = bt * 16 + lhi * 8;   // acc[r] -> batch row rowbase+r

  const float bias0 = biasbuf[0 * H_SZ + col];
  const float bias1 = biasbuf[1 * H_SZ + col];
  const float bias2 = biasbuf[2 * H_SZ + col];
  const float bias3 = biasbuf[3 * H_SZ + col];

  // wpack index: g*524288 + kb*16384 + nt*512 + lane*16
  const _Float16* wp_base = wpack + (size_t)nt * 512 + (size_t)lane * 16;

  for (int t = 0; t < T_STEPS; ++t) {
    const _Float16* __restrict__ hsrc = hbuf + (size_t)(t & 1) * (B_SZ * H_SZ);
    _Float16* __restrict__ hdst = hbuf + (size_t)((t + 1) & 1) * (B_SZ * H_SZ);

    v8f acc0, acc1, acc2, acc3;
#pragma unroll
    for (int r = 0; r < 8; ++r) {
      acc0[r] = bias0; acc1[r] = bias1; acc2[r] = bias2; acc3[r] = bias3;
    }

    // ---- K = 0..511 : x_t contribution (f32 -> f16 on the fly) ----
    const float* __restrict__ xrow = X + ((size_t)t * B_SZ + rg) * D_SZ;
#pragma unroll 4
    for (int kb = 0; kb < 16; ++kb) {
      const float* xp = xrow + kb * 32 + lhi * 8;
      const v8f xlo = *(const v8f*)(xp);
      const v8f xhi = *(const v8f*)(xp + 16);
      v16h a;
#pragma unroll
      for (int i = 0; i < 8; ++i) {
        a[i]     = (_Float16)xlo[i];
        a[8 + i] = (_Float16)xhi[i];
      }
      const _Float16* wp = wp_base + (size_t)kb * 16384;
      __builtin_prefetch(wp + 16384, 0, 1);
      const v16h b0 = *(const v16h*)(wp + 0u * 524288u);
      const v16h b1 = *(const v16h*)(wp + 1u * 524288u);
      const v16h b2 = *(const v16h*)(wp + 2u * 524288u);
      const v16h b3 = *(const v16h*)(wp + 3u * 524288u);
      acc0 = __builtin_amdgcn_wmma_f32_16x16x32_f16(false, a, false, b0,
                                                    (short)0, acc0, false, false);
      acc1 = __builtin_amdgcn_wmma_f32_16x16x32_f16(false, a, false, b1,
                                                    (short)0, acc1, false, false);
      acc2 = __builtin_amdgcn_wmma_f32_16x16x32_f16(false, a, false, b2,
                                                    (short)0, acc2, false, false);
      acc3 = __builtin_amdgcn_wmma_f32_16x16x32_f16(false, a, false, b3,
                                                    (short)0, acc3, false, false);
    }

    // ---- K = 512..1023 : h contribution (f16 direct) ----
    const _Float16* __restrict__ hrow = hsrc + (size_t)rg * H_SZ;
#pragma unroll 4
    for (int kb = 0; kb < 16; ++kb) {
      const _Float16* hp = hrow + kb * 32 + lhi * 8;
      const v8h hlo = *(const v8h*)(hp);
      const v8h hhi = *(const v8h*)(hp + 16);
      const v16h a = __builtin_shufflevector(hlo, hhi, 0, 1, 2, 3, 4, 5, 6, 7,
                                             8, 9, 10, 11, 12, 13, 14, 15);
      const _Float16* wp = wp_base + (size_t)(16 + kb) * 16384;
      __builtin_prefetch(wp + 16384, 0, 1);
      const v16h b0 = *(const v16h*)(wp + 0u * 524288u);
      const v16h b1 = *(const v16h*)(wp + 1u * 524288u);
      const v16h b2 = *(const v16h*)(wp + 2u * 524288u);
      const v16h b3 = *(const v16h*)(wp + 3u * 524288u);
      acc0 = __builtin_amdgcn_wmma_f32_16x16x32_f16(false, a, false, b0,
                                                    (short)0, acc0, false, false);
      acc1 = __builtin_amdgcn_wmma_f32_16x16x32_f16(false, a, false, b1,
                                                    (short)0, acc1, false, false);
      acc2 = __builtin_amdgcn_wmma_f32_16x16x32_f16(false, a, false, b2,
                                                    (short)0, acc2, false, false);
      acc3 = __builtin_amdgcn_wmma_f32_16x16x32_f16(false, a, false, b3,
                                                    (short)0, acc3, false, false);
    }

    // ---- gates + state update (each lane owns 8 disjoint (row,col) cells) --
#pragma unroll
    for (int r = 0; r < 8; ++r) {
      const int row = rowbase + r;
      const float iv = fast_sigmoid(acc0[r]);
      const float fv = fast_sigmoid(acc1[r]);
      const float ov = fast_sigmoid(acc2[r]);
      const float gv = fast_tanh(acc3[r]);
      const size_t sidx = (size_t)row * H_SZ + col;
      const float cold = Cstate[sidx];
      const float cnew = fv * cold + iv * gv;
      Cstate[sidx] = cnew;
      const float hv = ov * fast_tanh(cnew);
      out[((size_t)t * B_SZ + row) * H_SZ + col] = hv;
      hdst[sidx] = (_Float16)hv;
      if (t == T_STEPS - 1) Hout[sidx] = hv;
    }

    if (t + 1 < T_STEPS) grid_barrier(bar, (unsigned)t);
  }
}

extern "C" void kernel_launch(void* const* d_in, const int* in_sizes, int n_in,
                              void* d_out, int out_size, void* d_ws,
                              size_t ws_size, hipStream_t stream) {
  (void)in_sizes; (void)n_in; (void)out_size; (void)ws_size;
  const float* X   = (const float*)d_in[0];
  const float* Hin = (const float*)d_in[1];
  const float* Cin = (const float*)d_in[2];
  const float* Wxi = (const float*)d_in[3];
  const float* Whi = (const float*)d_in[4];
  const float* bi  = (const float*)d_in[5];
  const float* Wxf = (const float*)d_in[6];
  const float* Whf = (const float*)d_in[7];
  const float* bf  = (const float*)d_in[8];
  const float* Wxo = (const float*)d_in[9];
  const float* Who = (const float*)d_in[10];
  const float* bo  = (const float*)d_in[11];
  const float* Wxc = (const float*)d_in[12];
  const float* Whc = (const float*)d_in[13];
  const float* bc  = (const float*)d_in[14];

  char* ws = (char*)d_ws;
  _Float16* wpack  = (_Float16*)(ws + WS_WPACK_OFF);
  float*    biasb  = (float*)(ws + WS_BIAS_OFF);
  _Float16* hbuf   = (_Float16*)(ws + WS_HBUF_OFF);
  unsigned* bar    = (unsigned*)(ws + WS_BAR_OFF);

  float* out    = (float*)d_out;
  float* Hout   = out + (size_t)T_STEPS * B_SZ * H_SZ;
  float* Cstate = Hout + (size_t)B_SZ * H_SZ;  // running C lives in d_out

  const int NPACK = 4 * 32 * 32 * 32 * 16;
  pack_weights_kernel<<<(NPACK + 255) / 256, 256, 0, stream>>>(
      Wxi, Whi, bi, Wxf, Whf, bf, Wxo, Who, bo, Wxc, Whc, bc, wpack, biasb);
  prep_state_kernel<<<(B_SZ * H_SZ + 255) / 256, 256, 0, stream>>>(
      Hin, Cin, hbuf, Cstate, bar);
  lstm_persistent_kernel<<<NBLOCKS, NTHREADS, 0, stream>>>(
      X, wpack, biasb, hbuf, Cstate, out, Hout, bar);
}
